// SimplifiedAttention_25159918419980
// MI455X (gfx1250) — matrix operand order repaired and checked by reading, not compiled, not run on hardware
//
#include <hip/hip_runtime.h>

#define HEADS 8
#define HD    32
#define CDIM  256
#define NQ    16384
#define NKV   1024
#define BATCH 2
#define EPSV  1e-5f

typedef __bf16 bf16;
typedef bf16  bf16x16 __attribute__((ext_vector_type(16)));
typedef float f32x8   __attribute__((ext_vector_type(8)));

union Frag { bf16x16 v; uint4 u[2]; };

__device__ __forceinline__ unsigned short f2bf(float f) {
    unsigned int u = __builtin_bit_cast(unsigned int, f);
    u += 0x7fffu + ((u >> 16) & 1u);            // round-to-nearest-even
    return (unsigned short)(u >> 16);
}

// ---------------------------------------------------------------------------
// v[b,c] = mean over N of x[b,c,n]   (float4 streaming)
// ---------------------------------------------------------------------------
__global__ __launch_bounds__(256) void kmean(const float* __restrict__ x,
                                             float* __restrict__ v) {
    __shared__ float red[256];
    int row = blockIdx.x;                        // b*C + c
    const float4* xr = (const float4*)(x + (size_t)row * NQ);
    float s = 0.f;
    for (int i = threadIdx.x; i < NQ / 4; i += 256) {
        float4 t = xr[i];
        s += (t.x + t.y) + (t.z + t.w);
    }
    red[threadIdx.x] = s;
    __syncthreads();
    for (int off = 128; off > 0; off >>= 1) {
        if (threadIdx.x < off) red[threadIdx.x] += red[threadIdx.x + off];
        __syncthreads();
    }
    if (threadIdx.x == 0) v[row] = red[0] * (1.0f / NQ);
}

// ---------------------------------------------------------------------------
// Channel GEMM: out[b,o,n] = sum_c W[o,c] * X[b,c,n]
// Output packed bf16 as (b, h=o/32, n, d=o%32), repacked via LDS so global
// stores are coalesced b128.
// Block = 256 thr (8 waves): all 256 o x 64-wide n block (4 n-tiles).
// Wave w: o-tiles {2w,2w+1} x n-tiles {0..3} -> 8 accumulators.
// grid = dim3(ncols/64, B)
// ---------------------------------------------------------------------------
__global__ __launch_bounds__(256) void kchan(const float* __restrict__ W,
                                             const float* __restrict__ X,
                                             unsigned short* __restrict__ outbf,
                                             int ncols) {
    __shared__ unsigned short lw[CDIM * 32];        // [o][c_local]     16KB
    __shared__ unsigned short lx[64 * 32];          // [n_local][c_l]    4KB
    __shared__ unsigned short lout[HEADS * 64 * HD];// [h][n_local][d]  32KB
    int tid = threadIdx.x;
    int n0  = blockIdx.x * 64;
    int b   = blockIdx.y;
    int wave = tid >> 5, lane = tid & 31;
    int lm = lane & 15, lg = lane >> 4;

    f32x8 acc[2][4];
#pragma unroll
    for (int t = 0; t < 2; ++t)
#pragma unroll
        for (int j = 0; j < 4; ++j) acc[t][j] = (f32x8){0,0,0,0,0,0,0,0};

    for (int c0 = 0; c0 < CDIM; c0 += 32) {
        __syncthreads();
        // stage W[0:256][c0:c0+32] -> bf16 (coalesced: consecutive tid = consec c)
        for (int idx = tid; idx < CDIM * 32; idx += 256) {
            int o = idx >> 5, cl = idx & 31;
            lw[idx] = f2bf(W[o * CDIM + c0 + cl]);
        }
        // stage X[b][c0:+32][n0:+64] transposed to [n][c]
        for (int idx = tid; idx < 64 * 32; idx += 256) {
            int cl = idx >> 6, nl = idx & 63;
            lx[nl * 32 + cl] = f2bf(X[((size_t)(b * CDIM + c0 + cl)) * ncols + n0 + nl]);
        }
        // prefetch next K block
        if (c0 + 32 < CDIM) {
            __builtin_prefetch(&W[tid * CDIM + c0 + 32], 0, 1);
            if (tid < 32)
                __builtin_prefetch(&X[((size_t)(b * CDIM + c0 + 32 + tid)) * ncols + n0], 0, 1);
        }
        __syncthreads();

        // A fragments for this wave's two o-tiles
        Frag af[2];
#pragma unroll
        for (int t = 0; t < 2; ++t) {
            const uint4* lwq = (const uint4*)(lw + ((wave * 2 + t) * 16 + lm) * 32);
            af[t].u[0] = lwq[lg];                    // K = lg*8 .. +7
            af[t].u[1] = lwq[lg + 2];                // K = 16+lg*8 .. +7
        }
        const uint4* lxq = (const uint4*)lx;         // row = 32 ushorts = 4 uint4
#pragma unroll
        for (int j = 0; j < 4; ++j) {
            Frag bfr;
            bfr.u[0] = lxq[(j * 16 + lm) * 4 + lg];
            bfr.u[1] = lxq[(j * 16 + lm) * 4 + lg + 2];
#pragma unroll
            for (int t = 0; t < 2; ++t)
                acc[t][j] = __builtin_amdgcn_wmma_f32_16x16x32_bf16(
                    false, af[t].v, false, bfr.v, (short)0, acc[t][j], false, false);
        }
    }

    // repack D tiles into LDS as [h][n_local][d], then coalesced b128 stores
    __syncthreads();
#pragma unroll
    for (int t = 0; t < 2; ++t) {
        int ot = wave * 2 + t;
#pragma unroll
        for (int j = 0; j < 4; ++j) {
#pragma unroll
            for (int v = 0; v < 8; ++v) {
                int o  = ot * 16 + lg * 8 + v;
                int nl = j * 16 + lm;
                lout[((o >> 5) * 64 + nl) * HD + (o & 31)] = f2bf(acc[t][j][v]);
            }
        }
    }
    __syncthreads();
    const uint4* lo4 = (const uint4*)lout;
    uint4* ob4 = (uint4*)outbf;
    for (int idx = tid; idx < HEADS * 64 * HD / 8; idx += 256) {
        int h  = idx >> 8;                           // 256 uint4 per head
        int nl = (idx >> 2) & 63;
        int qo = idx & 3;
        ob4[(((size_t)(b * HEADS + h) * ncols) + n0 + nl) * 4 + qo] = lo4[idx];
    }
}

// ---------------------------------------------------------------------------
// SR conv (k=4,s=4) + BN: xr[b,o,m], K=(c,p,q)=4096.
// Block: all 256 o x 32-wide m block; wave: 2 o-tiles x 2 m-tiles.
// grid = dim3(NKV/32, B)
// ---------------------------------------------------------------------------
__global__ __launch_bounds__(256) void ksr(const float* __restrict__ x,
                                           const float* __restrict__ wsr,
                                           const float* __restrict__ g,
                                           const float* __restrict__ be,
                                           const float* __restrict__ mu,
                                           const float* __restrict__ var,
                                           float* __restrict__ xr) {
    __shared__ unsigned short lw[CDIM * 32];        // 16KB
    __shared__ unsigned short lx[32 * 32];          // [m_local][kk_local] 2KB
    int tid = threadIdx.x;
    int m0  = blockIdx.x * 32;
    int b   = blockIdx.y;
    int wave = tid >> 5, lane = tid & 31;
    int lm = lane & 15, lg = lane >> 4;
    int hh = m0 >> 5;                                // block spans one h' row

    f32x8 acc[2][2];
#pragma unroll
    for (int t = 0; t < 2; ++t)
#pragma unroll
        for (int j = 0; j < 2; ++j) acc[t][j] = (f32x8){0,0,0,0,0,0,0,0};

    for (int k0 = 0; k0 < CDIM * 16; k0 += 32) {
        __syncthreads();
        for (int idx = tid; idx < CDIM * 32; idx += 256) {
            int o = idx >> 5, kl = idx & 31;
            lw[idx] = f2bf(wsr[o * (CDIM * 16) + k0 + kl]);   // (o,c,p,q) contiguous
        }
        for (int idx = tid; idx < 32 * 32; idx += 256) {
            int kl = idx >> 5, ml = idx & 31;
            int kk = k0 + kl;
            int c = kk >> 4, pq = kk & 15, p = pq >> 2, q = pq & 3;
            lx[ml * 32 + kl] =
                f2bf(x[((size_t)(b * CDIM + c)) * NQ + (4 * hh + p) * 128 + 4 * ml + q]);
        }
        if (k0 + 32 < CDIM * 16)
            __builtin_prefetch(&wsr[tid * (CDIM * 16) + k0 + 32], 0, 1);
        __syncthreads();

        Frag af[2];
#pragma unroll
        for (int t = 0; t < 2; ++t) {
            const uint4* lwq = (const uint4*)(lw + ((wave * 2 + t) * 16 + lm) * 32);
            af[t].u[0] = lwq[lg];
            af[t].u[1] = lwq[lg + 2];
        }
        const uint4* lxq = (const uint4*)lx;
#pragma unroll
        for (int j = 0; j < 2; ++j) {
            Frag bfr;
            bfr.u[0] = lxq[(j * 16 + lm) * 4 + lg];
            bfr.u[1] = lxq[(j * 16 + lm) * 4 + lg + 2];
#pragma unroll
            for (int t = 0; t < 2; ++t)
                acc[t][j] = __builtin_amdgcn_wmma_f32_16x16x32_bf16(
                    false, af[t].v, false, bfr.v, (short)0, acc[t][j], false, false);
        }
    }
#pragma unroll
    for (int t = 0; t < 2; ++t) {
        int ot = wave * 2 + t;
#pragma unroll
        for (int j = 0; j < 2; ++j) {
#pragma unroll
            for (int v = 0; v < 8; ++v) {
                int o = ot * 16 + lg * 8 + v;
                int m = m0 + j * 16 + lm;
                float inv = g[o] * rsqrtf(var[o] + EPSV);
                xr[((size_t)(b * CDIM + o)) * NKV + m] = (acc[t][j][v] - mu[o]) * inv + be[o];
            }
        }
    }
}

// ---------------------------------------------------------------------------
// Attention max: attn[b,h,n] = scale * max_m <q[bh,n,:], k[bh,m,:]>
// Whole k[b,h] (64KB bf16) staged in LDS. Each wave: 2 n-tiles sharing the
// B fragment -> 2 WMMAs per B load. grid = dim3(NQ/256, B*HEADS), block 256.
// ---------------------------------------------------------------------------
__global__ __launch_bounds__(256) void kattn(const unsigned short* __restrict__ qbf,
                                             const unsigned short* __restrict__ kbf,
                                             float* __restrict__ attn) {
    __shared__ unsigned short lk[NKV * HD];          // 64KB: [m][d] bf16
    int tid = threadIdx.x;
    int bh  = blockIdx.y;
    const uint4* src = (const uint4*)(kbf + (size_t)bh * NKV * HD);
    uint4* dst = (uint4*)lk;
    for (int i = tid; i < NKV * HD / 8; i += 256) dst[i] = src[i];
    __syncthreads();

    int wave = tid >> 5, lane = tid & 31;
    int lm = lane & 15, lg = lane >> 4;
    int n0 = (blockIdx.x * 8 + wave) * 32;           // wave covers n0..n0+31

    Frag af[2];
#pragma unroll
    for (int t = 0; t < 2; ++t) {
        const uint4* qrow = (const uint4*)(qbf + ((size_t)bh * NQ + n0 + t * 16 + lm) * HD);
        af[t].u[0] = qrow[lg];
        af[t].u[1] = qrow[lg + 2];
    }

    f32x8 rmax[2];
    rmax[0] = (f32x8){-3.4e38f,-3.4e38f,-3.4e38f,-3.4e38f,
                      -3.4e38f,-3.4e38f,-3.4e38f,-3.4e38f};
    rmax[1] = rmax[0];

    const uint4* lkq = (const uint4*)lk;
    for (int j = 0; j < NKV / 16; ++j) {
        Frag bfr;
        bfr.u[0] = lkq[(j * 16 + lm) * 4 + lg];
        bfr.u[1] = lkq[(j * 16 + lm) * 4 + lg + 2];
#pragma unroll
        for (int t = 0; t < 2; ++t) {
            f32x8 c = {0,0,0,0,0,0,0,0};
            c = __builtin_amdgcn_wmma_f32_16x16x32_bf16(
                false, af[t].v, false, bfr.v, (short)0, c, false, false);
#pragma unroll
            for (int v = 0; v < 8; ++v) rmax[t][v] = fmaxf(rmax[t][v], c[v]);
        }
    }
    // reduce over the 16 kv columns (lanes within each 16-lane half)
#pragma unroll
    for (int t = 0; t < 2; ++t)
#pragma unroll
        for (int v = 0; v < 8; ++v) {
            float m = rmax[t][v];
            for (int d = 1; d < 16; d <<= 1) m = fmaxf(m, __shfl_xor(m, d, 32));
            rmax[t][v] = m;
        }
    const float scale = 0.17677669529663687f;        // 32^-0.5
    if (lm == 0) {
#pragma unroll
        for (int t = 0; t < 2; ++t)
#pragma unroll
            for (int v = 0; v < 8; ++v)
                attn[(size_t)bh * NQ + n0 + t * 16 + lg * 8 + v] = rmax[t][v] * scale;
    }
}

// ---------------------------------------------------------------------------
// s[b,n] = sum_h attn[b,h,n]
// ---------------------------------------------------------------------------
__global__ __launch_bounds__(256) void ksum(const float* __restrict__ attn,
                                            float* __restrict__ s) {
    int i = blockIdx.x * 256 + threadIdx.x;          // over B*NQ
    int b = i >> 14, n = i & (NQ - 1);
    float t = 0.f;
#pragma unroll
    for (int h = 0; h < HEADS; ++h) t += attn[((size_t)(b * HEADS + h)) * NQ + n];
    s[i] = t;
}

// ---------------------------------------------------------------------------
// pv[b,o] = sum_c w_proj[o,c] * v[b,c]
// ---------------------------------------------------------------------------
__global__ __launch_bounds__(256) void kpv(const float* __restrict__ wproj,
                                           const float* __restrict__ v,
                                           float* __restrict__ pv) {
    __shared__ float vs[CDIM];
    int b = blockIdx.x, o = threadIdx.x;
    vs[o] = v[b * CDIM + o];
    __syncthreads();
    float t = 0.f;
    for (int c = 0; c < CDIM; ++c) t += wproj[o * CDIM + c] * vs[c];
    pv[b * CDIM + o] = t;
}

// ---------------------------------------------------------------------------
// out[b,o,n] = BN( pv[b,o]*s[b,n] )    rank-1 outer product, float4 streaming
// ---------------------------------------------------------------------------
__global__ __launch_bounds__(256) void kout(const float* __restrict__ pv,
                                            const float* __restrict__ s,
                                            const float* __restrict__ g,
                                            const float* __restrict__ be,
                                            const float* __restrict__ mu,
                                            const float* __restrict__ var,
                                            float* __restrict__ out) {
    int bid = blockIdx.x;
    int chunk = bid & 15;
    int oc = bid >> 4;                               // b*C + o
    int o = oc & (CDIM - 1), b = oc >> 8;
    float inv = g[o] * rsqrtf(var[o] + EPSV);
    float pi  = pv[oc] * inv;
    float sh  = be[o] - mu[o] * inv;
    const float4* sv = (const float4*)(s + (size_t)b * NQ + chunk * 1024);
    float4* ov = (float4*)(out + (size_t)oc * NQ + chunk * 1024);
    float4 q = sv[threadIdx.x];
    float4 r;
    r.x = pi * q.x + sh;  r.y = pi * q.y + sh;
    r.z = pi * q.z + sh;  r.w = pi * q.w + sh;
    ov[threadIdx.x] = r;
}

// ---------------------------------------------------------------------------
extern "C" void kernel_launch(void* const* d_in, const int* in_sizes, int n_in,
                              void* d_out, int out_size, void* d_ws, size_t ws_size,
                              hipStream_t stream) {
    const float* x    = (const float*)d_in[0];
    const float* w_q  = (const float*)d_in[1];
    const float* w_k  = (const float*)d_in[2];
    const float* w_sr = (const float*)d_in[3];
    const float* sr_g = (const float*)d_in[4];
    const float* sr_b = (const float*)d_in[5];
    const float* sr_m = (const float*)d_in[6];
    const float* sr_v = (const float*)d_in[7];
    const float* w_pr = (const float*)d_in[8];
    const float* pr_g = (const float*)d_in[9];
    const float* pr_b = (const float*)d_in[10];
    const float* pr_m = (const float*)d_in[11];
    const float* pr_v = (const float*)d_in[12];
    float* out = (float*)d_out;

    char* ws = (char*)d_ws;
    unsigned short* qbf = (unsigned short*)(ws + 0);         // 16 MB
    unsigned short* kbf = (unsigned short*)(ws + 16777216);  // 1 MB
    float* xr    = (float*)(ws + 17825792);                  // 2 MB
    float* attn  = (float*)(ws + 19922944);                  // 1 MB
    float* sarr  = (float*)(ws + 20971520);                  // 128 KB
    float* vmean = (float*)(ws + 21102592);                  // 2 KB
    float* pvarr = (float*)(ws + 21104640);                  // 2 KB

    kmean<<<dim3(BATCH * CDIM), dim3(256), 0, stream>>>(x, vmean);
    kchan<<<dim3(NQ / 64, BATCH), dim3(256), 0, stream>>>(w_q, x, qbf, NQ);
    ksr<<<dim3(NKV / 32, BATCH), dim3(256), 0, stream>>>(x, w_sr, sr_g, sr_b, sr_m, sr_v, xr);
    kchan<<<dim3(NKV / 64, BATCH), dim3(256), 0, stream>>>(w_k, xr, kbf, NKV);
    kattn<<<dim3(NQ / 256, BATCH * HEADS), dim3(256), 0, stream>>>(qbf, kbf, attn);
    ksum<<<dim3(BATCH * NQ / 256), dim3(256), 0, stream>>>(attn, sarr);
    kpv<<<dim3(BATCH), dim3(256), 0, stream>>>(w_pr, vmean, pvarr);
    kout<<<dim3(BATCH * CDIM * 16), dim3(256), 0, stream>>>(pvarr, sarr, pr_g, pr_b, pr_m, pr_v, out);
}
